// Ernie4_5_DecoderLayer_82738249990876
// MI455X (gfx1250) — compile-verified, matrix operand order checked by reading, not run on hardware
//
#include <hip/hip_runtime.h>
#include <hip/hip_bf16.h>
#include <math.h>
#include <stdint.h>

// ---------------- types ----------------
typedef __bf16 bf16_t;
typedef bf16_t v16bf  __attribute__((ext_vector_type(16)));
typedef float  v8f    __attribute__((ext_vector_type(8)));
typedef bf16_t bf16x8 __attribute__((ext_vector_type(8)));
typedef bf16_t bf16x4 __attribute__((ext_vector_type(4)));

#define B_    2
#define S_    1024
#define SP_   1088      // padded key length for transposed V (covers causal overread, zero-filled)
#define D_    2048
#define T_    (B_*S_)
#define H_    16
#define HK_   4
#define HD_   128
#define E_    16
#define TOPK_ 6
#define I_    1024
#define IS_   2048

static __device__ __forceinline__ float  bf2f(bf16_t x) { return (float)x; }
static __device__ __forceinline__ bf16_t f2bf(float x)  { return (bf16_t)x; }

// A lane's 16 fragment elements (ISA 7.12.2, 16-bit A/B operands) are two
// contiguous 8-element runs along K: [kg*8, +8) and [16+kg*8, +8).
// If memory is K-contiguous at `row`, the whole fragment is two b128 loads.
static __device__ __forceinline__ v16bf frag_ld(const bf16_t* __restrict__ row, int kg) {
    union { v16bf v; bf16x8 h[2]; } u;
    u.h[0] = *reinterpret_cast<const bf16x8*>(row + kg * 8);
    u.h[1] = *reinterpret_cast<const bf16x8*>(row + 16 + kg * 8);
    return u.v;
}
static __device__ __forceinline__ v16bf frag_ld_lds(const bf16_t* row, int kg) {
    union { v16bf v; bf16x8 h[2]; } u;
    u.h[0] = *reinterpret_cast<const bf16x8*>(row + kg * 8);
    u.h[1] = *reinterpret_cast<const bf16x8*>(row + 16 + kg * 8);
    return u.v;
}

// gfx1250 async global->LDS copy (16B per lane), tracked by ASYNCcnt.
static __device__ __forceinline__ void async_copy_b128(uint32_t lds_addr, const void* gaddr) {
    asm volatile("global_load_async_to_lds_b128 %0, %1, off"
                 :: "v"(lds_addr), "v"((uint64_t)(uintptr_t)gaddr) : "memory");
}
static __device__ __forceinline__ void wait_async0() {
    asm volatile("s_wait_asynccnt 0x0" ::: "memory");
}

// ---------------- RMSNorm (fp32 in -> bf16 out) ----------------
__global__ __launch_bounds__(256)
void rmsnorm_bf16_kernel(const float* __restrict__ x, const float* __restrict__ w,
                         bf16_t* __restrict__ out, int Dm) {
    const int t = blockIdx.x;
    const float* xr = x + (size_t)t * Dm;
    bf16_t* orow = out + (size_t)t * Dm;
    float ss = 0.f;
    for (int d = threadIdx.x; d < Dm; d += blockDim.x) { float v = xr[d]; ss += v * v; }
    #pragma unroll
    for (int m = 16; m >= 1; m >>= 1) ss += __shfl_xor(ss, m, 32);
    __shared__ float red[8];
    if ((threadIdx.x & 31) == 0) red[threadIdx.x >> 5] = ss;
    __syncthreads();
    if (threadIdx.x == 0) {
        float v = 0.f;
        #pragma unroll
        for (int i = 0; i < 8; ++i) v += red[i];
        red[0] = rsqrtf(v / (float)Dm + 1e-6f);
    }
    __syncthreads();
    const float r = red[0];
    for (int d = threadIdx.x; d < Dm; d += blockDim.x)
        orow[d] = f2bf(w[d] * xr[d] * r);
}

// ---------------- generic WMMA GEMM (async A staging, double-buffered) ----------------
// Block tile 128x64x32, 8 waves, each wave owns 32x32 (4 WMMA tiles).
// A: bf16 row-major, staged via GLOBAL_LOAD_ASYNC_TO_LDS_B128 (ASYNCcnt).
// B: fp32 row-major -> bf16, staged K-major ([n][k]) so fragments are 2x ds_load_b128.
// C = base + rowscale * (A@B), base = Cin ? Cin : (accum ? C : 0).
// Requires M%128==0, N%64==0, K%32==0, K>=64.
__global__ __launch_bounds__(256)
void gemm_bf16w32_kernel(const bf16_t* __restrict__ A, int lda,
                         const float* __restrict__ Bw, int ldb,
                         float* __restrict__ C, int ldc,
                         const float* __restrict__ Cin,
                         const float* __restrict__ rs, int rs_stride,
                         int Mdim, int Ndim, int Kdim, int accum) {
    (void)Mdim; (void)Ndim;
    __shared__ __align__(16) bf16_t As[2][128][40];   // [buf][m][k], 80B rows: aligned, conflict-free
    __shared__ __align__(16) bf16_t Bs[2][64][40];    // [buf][n][k]
    const int tid  = threadIdx.x;
    const int wave = tid >> 5, lane = tid & 31;
    const int wm = wave & 3, wn = wave >> 2;          // wave grid 4(M) x 2(N)
    const int gm0 = blockIdx.y * 128;
    const int gn0 = blockIdx.x * 64;
    const int ml = lane & 15, kg = lane >> 4;

    // A staging: 2 chunks of 8 bf16 per thread (rows 0..63 and 64..127).
    const int r0 = tid >> 2, ac = (tid & 3) * 8;
    const int r1 = r0 + 64;
    const bf16_t* ap0 = A + (size_t)(gm0 + r0) * lda + ac;
    const bf16_t* ap1 = A + (size_t)(gm0 + r1) * lda + ac;
    const uint32_t alds0[2] = { (uint32_t)(uintptr_t)&As[0][r0][ac], (uint32_t)(uintptr_t)&As[1][r0][ac] };
    const uint32_t alds1[2] = { (uint32_t)(uintptr_t)&As[0][r1][ac], (uint32_t)(uintptr_t)&As[1][r1][ac] };
    // B staging: thread handles column n=tid&63, k-quarter kq=tid>>6 (8 strided dwords).
    const int bn = tid & 63, bkq = tid >> 6;
    const float* bp = Bw + (size_t)(bkq * 8) * ldb + gn0 + bn;
    bf16_t* const blds[2] = { &Bs[0][bn][bkq * 8], &Bs[1][bn][bkq * 8] };
    const size_t bstep = (size_t)32 * ldb;

    v8f acc[2][2];
    #pragma unroll
    for (int i = 0; i < 2; ++i)
        #pragma unroll
        for (int j = 0; j < 2; ++j) { v8f z = {}; acc[i][j] = z; }

    // stage tile 0 into buffer 0
    async_copy_b128(alds0[0], ap0);
    async_copy_b128(alds1[0], ap1);
    {
        bf16x8 bv;
        #pragma unroll
        for (int j = 0; j < 8; ++j) bv[j] = f2bf(bp[(size_t)j * ldb]);
        *reinterpret_cast<bf16x8*>(blds[0]) = bv;
    }

    const int nk = Kdim >> 5;
    for (int i = 0; i < nk; ++i) {
        const int buf = i & 1;
        wait_async0();            // own async copies for `buf` complete
        __syncthreads();          // everyone's copies/stores for `buf` visible; prior reads drained
        if (i + 1 < nk) {         // stage tile i+1 into buf^1, overlapped with this tile's WMMAs
            ap0 += 32; ap1 += 32; bp += bstep;
            async_copy_b128(alds0[buf ^ 1], ap0);
            async_copy_b128(alds1[buf ^ 1], ap1);
            bf16x8 bv;
            #pragma unroll
            for (int j = 0; j < 8; ++j) bv[j] = f2bf(bp[(size_t)j * ldb]);
            *reinterpret_cast<bf16x8*>(blds[buf ^ 1]) = bv;
        }
        v16bf af[2], bfv[2];
        #pragma unroll
        for (int mt = 0; mt < 2; ++mt)
            af[mt] = frag_ld_lds(&As[buf][(wm * 2 + mt) * 16 + ml][0], kg);
        #pragma unroll
        for (int nt = 0; nt < 2; ++nt)
            bfv[nt] = frag_ld_lds(&Bs[buf][(wn * 2 + nt) * 16 + ml][0], kg);
        #pragma unroll
        for (int mt = 0; mt < 2; ++mt)
            #pragma unroll
            for (int nt = 0; nt < 2; ++nt)
                acc[mt][nt] = __builtin_amdgcn_wmma_f32_16x16x32_bf16(
                    false, af[mt], false, bfv[nt], (short)0, acc[mt][nt], false, false);
    }
    // ---- epilogue ----
    #pragma unroll
    for (int mt = 0; mt < 2; ++mt) {
        #pragma unroll
        for (int nt = 0; nt < 2; ++nt) {
            #pragma unroll
            for (int r = 0; r < 8; ++r) {
                const int grow = gm0 + (wm * 2 + mt) * 16 + r + 8 * kg;
                const int gcol = gn0 + (wn * 2 + nt) * 16 + ml;
                const size_t off = (size_t)grow * ldc + gcol;
                float v = acc[mt][nt][r];
                if (rs) v *= rs[(size_t)grow * rs_stride];
                float base = 0.f;
                if (Cin) base = Cin[off];
                else if (accum) base = C[off];
                C[off] = base + v;
            }
        }
    }
}

// ---------------- RoPE: fp32 q/k -> bf16 rotated ----------------
__global__ void rope_kernel(const float* __restrict__ qf, const float* __restrict__ kf,
                            const float* __restrict__ cosb, const float* __restrict__ sinb,
                            bf16_t* __restrict__ qo, bf16_t* __restrict__ ko) {
    const int idx = blockIdx.x * blockDim.x + threadIdx.x;
    const int p = idx & 63;          // pair index (HD/2 = 64)
    const int row = idx >> 6;
    const int total_q = T_ * H_;
    if (row < total_q) {
        const int t = row / H_, h = row % H_;
        const size_t cs = (size_t)t * HD_ + 2 * p;       // t == b*S+s
        const float c = cosb[cs], s = sinb[cs];
        const size_t base = (size_t)t * (H_ * HD_) + (size_t)h * HD_ + 2 * p;
        const float x0 = qf[base], x1 = qf[base + 1];
        qo[base]     = f2bf(x0 * c - x1 * s);
        qo[base + 1] = f2bf(x1 * c + x0 * s);
    } else if (row < total_q + T_ * HK_) {
        const int row2 = row - total_q;
        const int t = row2 / HK_, h = row2 % HK_;
        const size_t cs = (size_t)t * HD_ + 2 * p;
        const float c = cosb[cs], s = sinb[cs];
        const size_t base = (size_t)t * (HK_ * HD_) + (size_t)h * HD_ + 2 * p;
        const float x0 = kf[base], x1 = kf[base + 1];
        ko[base]     = f2bf(x0 * c - x1 * s);
        ko[base + 1] = f2bf(x1 * c + x0 * s);
    }
}

// ---------------- V transpose: fp32 [t][kv*HD+d] -> bf16 [b,kv][d][key(SP_)] ----------------
// Key dim padded to SP_ and zero-filled so masked-key overreads contribute exact 0.
__global__ __launch_bounds__(256)
void transpose_v_kernel(const float* __restrict__ vf, bf16_t* __restrict__ vt) {
    __shared__ float tile[32][33];
    const int bk = blockIdx.z;                    // b*HK + kv
    const int d0 = blockIdx.x * 32;
    const int s0 = blockIdx.y * 32;               // padded key tile
    const int b = bk >> 2, kv = bk & 3;
    {
        const int kl = threadIdx.x >> 3, dc = (threadIdx.x & 7) * 4;
        const int key = s0 + kl;
        float4 v = make_float4(0.f, 0.f, 0.f, 0.f);
        if (key < S_)
            v = *reinterpret_cast<const float4*>(
                    vf + (size_t)(b * S_ + key) * (HK_ * HD_) + kv * HD_ + d0 + dc);
        tile[kl][dc] = v.x; tile[kl][dc + 1] = v.y; tile[kl][dc + 2] = v.z; tile[kl][dc + 3] = v.w;
    }
    __syncthreads();
    {
        const int dl = threadIdx.x >> 3, kc = (threadIdx.x & 7) * 4;
        bf16x4 o;
        o[0] = f2bf(tile[kc][dl]);     o[1] = f2bf(tile[kc + 1][dl]);
        o[2] = f2bf(tile[kc + 2][dl]); o[3] = f2bf(tile[kc + 3][dl]);
        *reinterpret_cast<bf16x4*>(vt + ((size_t)bk * HD_ + d0 + dl) * SP_ + s0 + kc) = o;
    }
}

// ---------------- silu(g) * u -> bf16 ----------------
__global__ void silu_mul_kernel(const float* __restrict__ g, const float* __restrict__ u,
                                bf16_t* __restrict__ out, int n) {
    const int i = blockIdx.x * blockDim.x + threadIdx.x;
    if (i < n) { const float x = g[i]; out[i] = f2bf((x / (1.f + __expf(-x))) * u[i]); }
}

// ---------------- WMMA flash attention (causal, GQA) ----------------
// 1 wave / 16-query tile; 32-key blocks. Q,K,V(T) fragments are all 2x b128
// global loads (K-contiguous); P bounces through LDS and reads back as b128.
__global__ __launch_bounds__(128)
void flash_attn_kernel(const bf16_t* __restrict__ Q, const bf16_t* __restrict__ K,
                       const bf16_t* __restrict__ Vt, bf16_t* __restrict__ O) {
    __shared__ bf16_t Pl[4][16][40];
    const int tid = threadIdx.x, wave = tid >> 5, lane = tid & 31;
    const int ml = lane & 15, kg = lane >> 4;
    const int h = blockIdx.y, b = blockIdx.z;
    const int q0 = (blockIdx.x * 4 + wave) * 16;
    const int kvh = h >> 2;                          // H/HK = 4
    const size_t qstr = H_ * HD_, kstr = HK_ * HD_;
    const bf16_t* Qb = Q + (size_t)(b * S_) * qstr + (size_t)h * HD_;
    const bf16_t* Kb = K + (size_t)(b * S_) * kstr + (size_t)kvh * HD_;
    const bf16_t* Vb = Vt + (size_t)(b * HK_ + kvh) * HD_ * SP_;   // [dim][key]

    v16bf qfr[4];
    #pragma unroll
    for (int kt = 0; kt < 4; ++kt)
        qfr[kt] = frag_ld(Qb + (size_t)(q0 + ml) * qstr + kt * 32, kg);

    float mi[8], li[8]; v8f o[8];
    #pragma unroll
    for (int r = 0; r < 8; ++r) { mi[r] = -1e30f; li[r] = 0.f; }
    #pragma unroll
    for (int ct = 0; ct < 8; ++ct) { v8f z = {}; o[ct] = z; }

    const float scale = 0.0883883476483184f;         // 128^-0.5
    const int jend = q0 + 16;
    for (int j0 = 0; j0 < jend; j0 += 32) {
        v8f s0 = {}, s1 = {};
        const int key0 = j0 + ml, key1 = j0 + 16 + ml;
        const int kc0 = key0 < S_ - 1 ? key0 : S_ - 1;
        const int kc1 = key1 < S_ - 1 ? key1 : S_ - 1;
        const bf16_t* kr0 = Kb + (size_t)kc0 * kstr;
        const bf16_t* kr1 = Kb + (size_t)kc1 * kstr;
        #pragma unroll
        for (int kt = 0; kt < 4; ++kt) {
            v16bf k0f = frag_ld(kr0 + kt * 32, kg);
            v16bf k1f = frag_ld(kr1 + kt * 32, kg);
            s0 = __builtin_amdgcn_wmma_f32_16x16x32_bf16(false, qfr[kt], false, k0f, (short)0, s0, false, false);
            s1 = __builtin_amdgcn_wmma_f32_16x16x32_bf16(false, qfr[kt], false, k1f, (short)0, s1, false, false);
        }
        #pragma unroll
        for (int r = 0; r < 8; ++r) {
            const int qrow = q0 + r + 8 * kg;
            float v0 = s0[r] * scale + ((key0 <= qrow) ? 0.f : -1e30f);
            float v1 = s1[r] * scale + ((key1 <= qrow) ? 0.f : -1e30f);
            float mx = fmaxf(v0, v1);
            #pragma unroll
            for (int msk = 1; msk < 16; msk <<= 1) mx = fmaxf(mx, __shfl_xor(mx, msk, 32));
            const float mnew = fmaxf(mi[r], mx);
            const float corr = __expf(mi[r] - mnew);
            const float p0 = __expf(v0 - mnew), p1 = __expf(v1 - mnew);
            float rsum = p0 + p1;
            #pragma unroll
            for (int msk = 1; msk < 16; msk <<= 1) rsum += __shfl_xor(rsum, msk, 32);
            li[r] = li[r] * corr + rsum; mi[r] = mnew;
            #pragma unroll
            for (int ct = 0; ct < 8; ++ct) o[ct][r] *= corr;
            Pl[wave][r + 8 * kg][ml]      = f2bf(p0);
            Pl[wave][r + 8 * kg][16 + ml] = f2bf(p1);
        }
        v16bf pfr = frag_ld_lds(&Pl[wave][ml][0], kg);
        #pragma unroll
        for (int ct = 0; ct < 8; ++ct) {
            v16bf vfr = frag_ld(Vb + (size_t)(ct * 16 + ml) * SP_ + j0, kg);
            o[ct] = __builtin_amdgcn_wmma_f32_16x16x32_bf16(false, pfr, false, vfr, (short)0, o[ct], false, false);
        }
    }
    bf16_t* Ob = O + (size_t)(b * S_) * qstr + (size_t)h * HD_;
    #pragma unroll
    for (int r = 0; r < 8; ++r) {
        const int qrow = q0 + r + 8 * kg;
        const float inv = 1.f / li[r];
        #pragma unroll
        for (int ct = 0; ct < 8; ++ct)
            Ob[(size_t)qrow * qstr + ct * 16 + ml] = f2bf(o[ct][r] * inv);
    }
}

// ---------------- router: softmax + biased top-6 + renorm ----------------
__global__ __launch_bounds__(32)
void route_kernel(const bf16_t* __restrict__ xn, const float* __restrict__ Wgate,
                  const float* __restrict__ bias, float* __restrict__ route) {
    const int t = blockIdx.x, lane = threadIdx.x;
    __shared__ float lg[E_];
    if (lane < E_) {
        const bf16_t* xr = xn + (size_t)t * D_;
        float acc = 0.f;
        for (int d = 0; d < D_; ++d) acc += bf2f(xr[d]) * Wgate[(size_t)d * E_ + lane];
        lg[lane] = acc;
    }
    __syncthreads();
    if (lane == 0) {
        float mx = -1e30f;
        for (int e = 0; e < E_; ++e) mx = fmaxf(mx, lg[e]);
        float p[E_], sum = 0.f;
        for (int e = 0; e < E_; ++e) { p[e] = __expf(lg[e] - mx); sum += p[e]; }
        for (int e = 0; e < E_; ++e) p[e] /= sum;
        bool sel[E_]; for (int e = 0; e < E_; ++e) sel[e] = false;
        int selidx[TOPK_]; float wsum = 0.f;
        for (int k = 0; k < TOPK_; ++k) {
            int best = 0; float bv = -1e30f;
            for (int e = 0; e < E_; ++e)
                if (!sel[e]) { const float v = p[e] + bias[e]; if (v > bv) { bv = v; best = e; } }
            sel[best] = true; selidx[k] = best; wsum += p[best];
        }
        wsum = fmaxf(wsum, 1e-12f);
        float outw[E_]; for (int e = 0; e < E_; ++e) outw[e] = 0.f;
        for (int k = 0; k < TOPK_; ++k) outw[selidx[k]] = p[selidx[k]] / wsum;
        for (int e = 0; e < E_; ++e) route[(size_t)t * E_ + e] = outw[e];
    }
}

// ---------------- host orchestration ----------------
extern "C" void kernel_launch(void* const* d_in, const int* in_sizes, int n_in,
                              void* d_out, int out_size, void* d_ws, size_t ws_size,
                              hipStream_t stream) {
    (void)in_sizes; (void)n_in; (void)out_size; (void)ws_size;
    const float* hidden = (const float*)d_in[0];
    const float* cosb   = (const float*)d_in[1];
    const float* sinb   = (const float*)d_in[2];
    const float* ln1w   = (const float*)d_in[3];
    const float* ln2w   = (const float*)d_in[4];
    const float* Wq     = (const float*)d_in[5];
    const float* Wk     = (const float*)d_in[6];
    const float* Wv     = (const float*)d_in[7];
    const float* Wo     = (const float*)d_in[8];
    const float* Wgate  = (const float*)d_in[9];
    const float* cbias  = (const float*)d_in[10];
    const float* Wg     = (const float*)d_in[11];
    const float* Wu     = (const float*)d_in[12];
    const float* Wd     = (const float*)d_in[13];
    const float* Wgs    = (const float*)d_in[14];
    const float* Wus    = (const float*)d_in[15];
    const float* Wds    = (const float*)d_in[16];
    float* out = (float*)d_out;

    char* p = (char*)d_ws;
    auto alloc = [&](size_t bytes) { void* r = (void*)p; p += (bytes + 255) & ~(size_t)255; return r; };
    bf16_t* h1n  = (bf16_t*)alloc((size_t)T_ * D_ * 2);
    float*  qf   = (float*) alloc((size_t)T_ * H_  * HD_ * 4);
    float*  kf   = (float*) alloc((size_t)T_ * HK_ * HD_ * 4);
    float*  vf   = (float*) alloc((size_t)T_ * HK_ * HD_ * 4);
    bf16_t* qb   = (bf16_t*)alloc((size_t)T_ * H_  * HD_ * 2);
    bf16_t* kb   = (bf16_t*)alloc((size_t)T_ * HK_ * HD_ * 2);
    bf16_t* vt   = (bf16_t*)alloc((size_t)B_ * HK_ * HD_ * SP_ * 2);
    bf16_t* ao   = (bf16_t*)alloc((size_t)T_ * H_  * HD_ * 2);
    float*  r2   = (float*) alloc((size_t)T_ * D_ * 4);
    bf16_t* h2n  = (bf16_t*)alloc((size_t)T_ * D_ * 2);
    float*  route= (float*) alloc((size_t)T_ * E_ * 4);
    float*  gbuf = (float*) alloc((size_t)T_ * IS_ * 4);
    float*  ubuf = (float*) alloc((size_t)T_ * IS_ * 4);
    bf16_t* hb   = (bf16_t*)alloc((size_t)T_ * IS_ * 2);

    auto gemm = [&](const bf16_t* A, int lda, const float* Bw, int ldb,
                    float* C, int ldc, const float* Cin,
                    const float* rs, int rss, int N, int K, int accum) {
        dim3 g(N / 64, T_ / 128);
        gemm_bf16w32_kernel<<<g, 256, 0, stream>>>(A, lda, Bw, ldb, C, ldc, Cin, rs, rss, T_, N, K, accum);
    };

    // ---- attention block ----
    rmsnorm_bf16_kernel<<<T_, 256, 0, stream>>>(hidden, ln1w, h1n, D_);
    gemm(h1n, D_, Wq, H_ * HD_,  qf, H_ * HD_,  nullptr, nullptr, 0, H_ * HD_,  D_, 0);
    gemm(h1n, D_, Wk, HK_ * HD_, kf, HK_ * HD_, nullptr, nullptr, 0, HK_ * HD_, D_, 0);
    gemm(h1n, D_, Wv, HK_ * HD_, vf, HK_ * HD_, nullptr, nullptr, 0, HK_ * HD_, D_, 0);
    {
        int nr = T_ * (H_ + HK_) * (HD_ / 2);
        rope_kernel<<<(nr + 255) / 256, 256, 0, stream>>>(qf, kf, cosb, sinb, qb, kb);
        transpose_v_kernel<<<dim3(HD_ / 32, SP_ / 32, B_ * HK_), 256, 0, stream>>>(vf, vt);
    }
    flash_attn_kernel<<<dim3(S_ / 64, H_, B_), 128, 0, stream>>>(qb, kb, vt, ao);
    // resid2 = hidden + attn_out @ Wo
    gemm(ao, H_ * HD_, Wo, D_, r2, D_, hidden, nullptr, 0, D_, H_ * HD_, 0);

    // ---- MoE block ----
    rmsnorm_bf16_kernel<<<T_, 256, 0, stream>>>(r2, ln2w, h2n, D_);
    route_kernel<<<T_, 32, 0, stream>>>(h2n, Wgate, cbias, route);

    // shared expert: out = resid2 + (silu(h2n@Wgs) * (h2n@Wus)) @ Wds
    gemm(h2n, D_, Wgs, IS_, gbuf, IS_, nullptr, nullptr, 0, IS_, D_, 0);
    gemm(h2n, D_, Wus, IS_, ubuf, IS_, nullptr, nullptr, 0, IS_, D_, 0);
    {
        int n = T_ * IS_;
        silu_mul_kernel<<<(n + 255) / 256, 256, 0, stream>>>(gbuf, ubuf, hb, n);
    }
    gemm(hb, IS_, Wds, D_, out, D_, r2, nullptr, 0, D_, IS_, 0);

    // routed experts (dense over all E; route weight is 0 for unselected -> exact semantics)
    for (int e = 0; e < E_; ++e) {
        const float* Wge = Wg + (size_t)e * D_ * I_;
        const float* Wue = Wu + (size_t)e * D_ * I_;
        const float* Wde = Wd + (size_t)e * I_ * D_;
        gemm(h2n, D_, Wge, I_, gbuf, I_, nullptr, nullptr, 0, I_, D_, 0);
        gemm(h2n, D_, Wue, I_, ubuf, I_, nullptr, nullptr, 0, I_, D_, 0);
        int n = T_ * I_;
        silu_mul_kernel<<<(n + 255) / 256, 256, 0, stream>>>(gbuf, ubuf, hb, n);
        gemm(hb, I_, Wde, D_, out, D_, nullptr, route + e, E_, D_, I_, 1);
    }
}
